// SelfAttention_16630113370502
// MI455X (gfx1250) — matrix operand order repaired and checked by reading, not compiled, not run on hardware
//
#include <hip/hip_runtime.h>
#include <hip/hip_bf16.h>
#include <stdint.h>

// Problem constants (fixed by the reference harness)
constexpr int B_   = 4;
constexpr int C_   = 256;      // channels
constexpr int T_   = 4096;     // sequence length
constexpr int C8_  = 32;       // q/k channels
constexpr int NSC_ = T_ / 32;  // 32-wide s-chunks (pass B)
constexpr int NCA_ = T_ / 128; // 128-wide s-chunks (pass A)

constexpr float LOG2E_ = 1.4426950408889634f;

typedef __attribute__((ext_vector_type(16))) __bf16        v16bf;
typedef __attribute__((ext_vector_type(8)))  float         v8f;
typedef __attribute__((ext_vector_type(8)))  unsigned int  v8u;

static __device__ __forceinline__ unsigned short f2bf(float f) {
    unsigned int u = __builtin_bit_cast(unsigned int, f);
    unsigned int r = (u + 0x7FFFu + ((u >> 16) & 1u)) >> 16; // RNE
    return (unsigned short)r;
}
#if __has_builtin(__builtin_amdgcn_cvt_pk_bf16_f32)
typedef __attribute__((ext_vector_type(2))) __bf16 v2bf;
static __device__ __forceinline__ unsigned int pack2(float lo, float hi) {
    v2bf p = __builtin_amdgcn_cvt_pk_bf16_f32(lo, hi);
    return __builtin_bit_cast(unsigned int, p);
}
#else
static __device__ __forceinline__ unsigned int pack2(float lo, float hi) {
    return (unsigned int)f2bf(lo) | ((unsigned int)f2bf(hi) << 16);
}
#endif
static __device__ __forceinline__ v16bf as_bf(v8u u) {
    return __builtin_bit_cast(v16bf, u);
}

// ---- CDNA5 async global->LDS copy (ASYNCcnt-tracked) ------------------------
#define ASYNC_B128(ldsoff, gaddr)                                              \
    asm volatile("global_load_async_to_lds_b128 %0, %1, off"                   \
                 :: "v"(ldsoff), "v"(gaddr) : "memory")

#if __has_builtin(__builtin_amdgcn_s_wait_asynccnt)
#define WAIT_ASYNC(n) __builtin_amdgcn_s_wait_asynccnt(n)
#else
#define WAIT_ASYNC(n) asm volatile("s_wait_asynccnt %0" :: "n"(n) : "memory")
#endif

// LDS arena layout (bytes). Bank-conflict padding is applied by the async
// copies themselves (each lane transfer carries its own LDS destination):
//   k rows: 64B data on an 80B stride  (dword stride 20 -> 16 distinct banks)
//   v 32B blocks on a 48B stride       (2-way instead of 4-way, 16B aligned)
constexpr int KROW_ = 80;                 // padded k row stride
constexpr int V32_  = 48;                 // padded v 32B-block stride
constexpr int KSZ_  = 32 * KROW_;         // 2560  (pass-B k buffer)
constexpr int VSZ_  = 512 * V32_;         // 24576 (pass-B v buffer / pass-A k: 128 rows x 80)
constexpr int VS0_  = 2 * KSZ_;
constexpr int LBUF_ = VS0_ + 2 * VSZ_;    // 4 waves x 16 floats of 1/l
constexpr int SMEM_BYTES = LBUF_ + 4 * 16 * 4;

// ---------------------------------------------------------------------------
// Kernel 1: fused q/k/v projections.  qb[b][t][c], kb[b][s][c] row-major bf16;
// vpk pre-swizzled so a P*V B-matrix tile is 8 contiguous dwords per lane.
// q is pre-scaled by log2(e) so pass B can use exp2 directly (no v_mul per exp).
// Two wave-uniform output rows per x read (2 FMA per ds_load).
// ---------------------------------------------------------------------------
__global__ __launch_bounds__(256) void proj_kernel(
    const float* __restrict__ x,
    const float* __restrict__ Wq, const float* __restrict__ bq,
    const float* __restrict__ Wk, const float* __restrict__ bk,
    const float* __restrict__ Wv, const float* __restrict__ bv,
    unsigned short* __restrict__ qb,
    unsigned short* __restrict__ kb,
    unsigned short* __restrict__ vpk)
{
    __shared__ float xs[C_ * 64];  // 64 KB x-tile: [c][t_local]
    const int b  = blockIdx.y;
    const int t0 = blockIdx.x * 64;
    const int tid = threadIdx.x;

    for (int i = tid; i < C_ * 64; i += 256) {
        const int c = i >> 6, tl = i & 63;
        xs[i] = x[((size_t)b * C_ + c) * T_ + t0 + tl];
    }
    __syncthreads();

    const int tl  = tid & 63;
    const int cog = tid >> 6;         // uniform within each wave32
    const int t   = t0 + tl;

    auto wsel = [&](int co, const float*& W, float& bias) {
        if (co < C8_)        { W = Wq + (size_t)co * C_;           bias = bq[co]; }
        else if (co < 2*C8_) { W = Wk + (size_t)(co - C8_) * C_;   bias = bk[co - C8_]; }
        else                 { W = Wv + (size_t)(co - 2*C8_) * C_; bias = bv[co - 2*C8_]; }
    };
    auto emit = [&](int co, float acc) {
        if (co < C8_) acc *= LOG2E_;     // fold exp->exp2 conversion into q
        const unsigned short hb = f2bf(acc);
        if (co < C8_) {
            qb[((size_t)b * T_ + t) * C8_ + co] = hb;
        } else if (co < 2 * C8_) {
            kb[((size_t)b * T_ + t) * C8_ + (co - C8_)] = hb;
        } else {
            const int cc = co - 2 * C8_;
            const int sc = t >> 5, r = t & 31;
            const int hh = (r >> 3) & 1, s2 = (r & 7) | ((r >> 4) << 3), j = s2 >> 1, par = s2 & 1;
            const size_t d = ((((size_t)b * NSC_ + sc) * C_ + cc) * 2 + hh) * 8 + j;
            vpk[d * 2 + par] = hb;
        }
    };

    for (int co = cog; co < (2 * C8_ + C_); co += 8) {
        const int co2 = co + 4;                 // also wave-uniform
        const float *W1, *W2; float b1, b2;
        wsel(co,  W1, b1);
        wsel(co2, W2, b2);
        float a1 = b1, a2 = b2;
        #pragma unroll 8
        for (int c = 0; c < C_; ++c) {
            const float xv = xs[c * 64 + tl];
            a1 = fmaf(W1[c], xv, a1);
            a2 = fmaf(W2[c], xv, a2);
        }
        emit(co,  a1);
        emit(co2, a2);
    }
}

// ---------------------------------------------------------------------------
// Kernel 2: flash attention. Block = 4 wave32 = 4 query tiles (same batch).
// Pass A: row-max only (128-s chunks, max3 chains, no exp).
// Pass B: p = exp2(e - mt) unscaled (q pre-scaled by log2 e), l accumulated
//         from the same p values, O += P x V.  1/l applied in the epilogue.
// Epilogue: per-wave 16x16 tile transpose through LDS -> float4-coalesced
//           residual read + output write along t.
// ---------------------------------------------------------------------------
__global__ __launch_bounds__(128) void attn_kernel(
    const unsigned int* __restrict__ qb,
    const unsigned int* __restrict__ kbuf,
    const unsigned int* __restrict__ vpk,
    const float* __restrict__ x,
    const float* __restrict__ gamma,
    float* __restrict__ out)
{
    __shared__ __align__(16) char smem[SMEM_BYTES];

    const int lane = threadIdx.x & 31;
    const int wv   = threadIdx.x >> 5;
    const int tile = blockIdx.x * 4 + wv;        // B*T/16 = 1024 tiles
    const int b    = tile >> 8;                  // 256 tiles per batch
    const int t0   = (tile & 255) << 4;
    const int tcol = lane & 15;
    const int hf   = lane >> 4;

    const v8f zero = {};

    // q tile (B-matrix layout): lane -> t = t0+tcol; dwords 8*hf .. 8*hf+7
    const unsigned int* qrow = qb + ((size_t)b * T_ + t0 + tcol) * (C8_ / 2);
    v8u qreg;
    {
        uint4 a = *(const uint4*)(qrow + 8 * hf);
        uint4 c = *(const uint4*)(qrow + 8 * hf + 4);
        qreg = (v8u){a.x, a.y, a.z, a.w, c.x, c.y, c.z, c.w};
    }
    const v16bf qv = as_bf(qreg);

    const char* kg = (const char*)kbuf + (size_t)b * T_ * 64;          // 64 B per k row
    const char* vg = (const char*)vpk  + (size_t)b * NSC_ * (C_ * 64); // 16 KB per sc
    const unsigned smem_base = (unsigned)(size_t)&smem[0];
    const unsigned ks_off[2] = { 0u, (unsigned)KSZ_ };
    const unsigned vs_off[2] = { (unsigned)VS0_, (unsigned)(VS0_ + VSZ_) };

    // Loop-invariant per-lane copy mappings (padding applied on the LDS side)
    unsigned kB_src, kB_dst;
    { const unsigned off = wv * 512 + lane * 16, ch = off >> 4;
      kB_src = off; kB_dst = (ch >> 2) * KROW_ + ((ch & 3) << 4); }
    unsigned kA_src[4], kA_dst[4];
    #pragma unroll
    for (int j = 0; j < 4; ++j) {
        const unsigned off = wv * 2048 + j * 512 + lane * 16, ch = off >> 4;
        kA_src[j] = off; kA_dst[j] = (ch >> 2) * KROW_ + ((ch & 3) << 4);
    }
    unsigned vB_src[8], vB_dst[8];
    #pragma unroll
    for (int j = 0; j < 8; ++j) {
        const unsigned off = wv * 4096 + j * 512 + lane * 16, ch = off >> 4;
        vB_src[j] = off; vB_dst[j] = (ch >> 1) * V32_ + ((ch & 1) << 4);
    }

    auto issueA = [&](int buf, int ci) {     // 8 KB of k rows -> vs area
        #pragma unroll
        for (int j = 0; j < 4; ++j)
            ASYNC_B128(smem_base + vs_off[buf] + kA_dst[j],
                       kg + (size_t)ci * 8192 + kA_src[j]);
    };
    auto issue_k = [&](int buf, int ci) {    // 2 KB of k rows
        ASYNC_B128(smem_base + ks_off[buf] + kB_dst, kg + (size_t)ci * 2048 + kB_src);
    };
    auto issue_v = [&](int buf, int ci) {    // 16 KB v chunk
        #pragma unroll
        for (int j = 0; j < 8; ++j)
            ASYNC_B128(smem_base + vs_off[buf] + vB_dst[j],
                       vg + (size_t)ci * 16384 + vB_src[j]);
    };
    // k tile (A-matrix layout) from a padded-row LDS region
    auto ktile = [&](unsigned region, int u) -> v16bf {
        const char* base = smem + region + (unsigned)((u * 16 + tcol) * KROW_);
        uint4 a = *(const uint4*)(base + 32 * hf);
        uint4 c = *(const uint4*)(base + 32 * hf + 16);
        v8u r = (v8u){a.x, a.y, a.z, a.w, c.x, c.y, c.z, c.w};
        return as_bf(r);
    };

    // ---- Pass A: row max only, 128-s chunks through the v buffers ----------
    float m = -3.0e38f;
    issueA(0, 0);
    issueA(1, 1);
    for (int ci = 0; ci < NCA_; ++ci) {
        const int buf = ci & 1;
        if (ci + 1 < NCA_) { WAIT_ASYNC(4); } else { WAIT_ASYNC(0); }
        __syncthreads();
        #pragma unroll
        for (int u = 0; u < 8; ++u) {
            v16bf ka = ktile(vs_off[buf], u);
            v8f e = __builtin_amdgcn_wmma_f32_16x16x32_bf16(
                        false, ka, false, qv, (short)0, zero, false, false);
            #pragma unroll
            for (int r = 0; r < 8; ++r) m = fmaxf(m, e[r]);
        }
        __syncthreads();
        if (ci + 2 < NCA_) issueA(buf, ci + 2);
    }
    const float mt = fmaxf(m, __shfl_xor(m, 16, 32));

    // ---- Pass B: O accumulation + l from the same p values -----------------
    v8f O[16];
    #pragma unroll
    for (int i = 0; i < 16; ++i) O[i] = zero;
    float l = 0.0f;

    issue_k(0, 0); issue_v(0, 0);
    issue_k(1, 1); issue_v(1, 1);
    for (int ci = 0; ci < NSC_; ++ci) {
        const int buf = ci & 1;
        if (ci + 1 < NSC_) { WAIT_ASYNC(9); } else { WAIT_ASYNC(0); }
        __syncthreads();

        v16bf ka0 = ktile(ks_off[buf], 0);
        v16bf ka1 = ktile(ks_off[buf], 1);
        v8f e0 = __builtin_amdgcn_wmma_f32_16x16x32_bf16(
                     false, ka0, false, qv, (short)0, zero, false, false);
        v8f e1 = __builtin_amdgcn_wmma_f32_16x16x32_bf16(
                     false, ka1, false, qv, (short)0, zero, false, false);

        float p0[8], p1[8], ps = 0.0f;
        #pragma unroll
        for (int r = 0; r < 8; ++r) { p0[r] = exp2f(e0[r] - mt); ps += p0[r]; }
        #pragma unroll
        for (int r = 0; r < 8; ++r) { p1[r] = exp2f(e1[r] - mt); ps += p1[r]; }
        l += ps;

        v8u ap;
        #pragma unroll
        for (int j = 0; j < 4; ++j) {
            ap[j]     = pack2(p0[2*j], p0[2*j+1]);
            ap[4 + j] = pack2(p1[2*j], p1[2*j+1]);
        }
        const v16bf Ap = as_bf(ap);

        #pragma unroll
        for (int ct = 0; ct < 16; ++ct) {
            const char* vb = smem + vs_off[buf]
                           + (unsigned)((((ct * 16 + tcol) * 2) + hf) * V32_);
            uint4 a = *(const uint4*)(vb);
            uint4 c = *(const uint4*)(vb + 16);
            v8u vr = (v8u){a.x, a.y, a.z, a.w, c.x, c.y, c.z, c.w};
            O[ct] = __builtin_amdgcn_wmma_f32_16x16x32_bf16(
                        false, Ap, false, as_bf(vr), (short)0, O[ct], false, false);
        }
        __syncthreads();
        if (ci + 2 < NSC_) { issue_k(buf, ci + 2); issue_v(buf, ci + 2); }
    }

    // ---- 1/l per t, broadcast through LDS (t varies across VGPRs in-lane)
    const float lt = l + __shfl_xor(l, 16, 32);
    const float rl = 1.0f / lt;
    float* lbufp = (float*)(smem + LBUF_ + wv * 64);
    if (lane < 16) lbufp[lane] = rl;     // lane -> t0 + lane

    // ---- Epilogue: transpose O tiles through LDS, float4-coalesced along t
    const float g = gamma[0];
    float* tilep = (float*)(smem + wv * 1024);   // per-wave 16x16 f32 staging
    const int cr = lane >> 1;
    const int tq = (lane & 1) * 8;
    const float4 rl0 = *(const float4*)(lbufp + tq);
    const float4 rl1 = *(const float4*)(lbufp + tq + 4);
    #pragma unroll
    for (int ct = 0; ct < 16; ++ct) {
        // stage D tile: element (c=tcol, t=r+8*hf) -> tilep[c*16 + t]
        float4 lo = { O[ct][0], O[ct][1], O[ct][2], O[ct][3] };
        float4 hi = { O[ct][4], O[ct][5], O[ct][6], O[ct][7] };
        *(float4*)(tilep + tcol * 16 + 8 * hf)     = lo;
        *(float4*)(tilep + tcol * 16 + 8 * hf + 4) = hi;
        // read back transposed: lane -> row c = cr, t = t0 + tq .. +7
        const float4 o0 = *(const float4*)(tilep + cr * 16 + tq);
        const float4 o1 = *(const float4*)(tilep + cr * 16 + tq + 4);
        const size_t idx = ((size_t)b * C_ + ct * 16 + cr) * (size_t)T_ + t0 + tq;
        const float4 x0 = *(const float4*)(x + idx);
        const float4 x1 = *(const float4*)(x + idx + 4);
        float4 r0, r1;
        r0.x = g * o0.x * rl0.x + x0.x;  r0.y = g * o0.y * rl0.y + x0.y;
        r0.z = g * o0.z * rl0.z + x0.z;  r0.w = g * o0.w * rl0.w + x0.w;
        r1.x = g * o1.x * rl1.x + x1.x;  r1.y = g * o1.y * rl1.y + x1.y;
        r1.z = g * o1.z * rl1.z + x1.z;  r1.w = g * o1.w * rl1.w + x1.w;
        *(float4*)(out + idx)     = r0;
        *(float4*)(out + idx + 4) = r1;
    }
}

// ---------------------------------------------------------------------------
extern "C" void kernel_launch(void* const* d_in, const int* in_sizes, int n_in,
                              void* d_out, int out_size, void* d_ws, size_t ws_size,
                              hipStream_t stream)
{
    const float* x     = (const float*)d_in[0];
    const float* Wq    = (const float*)d_in[1];
    const float* bq    = (const float*)d_in[2];
    const float* Wk    = (const float*)d_in[3];
    const float* bk    = (const float*)d_in[4];
    const float* Wv    = (const float*)d_in[5];
    const float* bv    = (const float*)d_in[6];
    const float* gamma = (const float*)d_in[7];
    float* out = (float*)d_out;

    // Workspace carve-up (bf16): qb 1 MB, kb 1 MB, vpk 8 MB  (total 10 MB)
    char* ws = (char*)d_ws;
    unsigned short* qb  = (unsigned short*)(ws);
    unsigned short* kb  = (unsigned short*)(ws + (1u << 20));
    unsigned short* vpk = (unsigned short*)(ws + (2u << 20));

    dim3 pgrid(T_ / 64, B_);
    proj_kernel<<<pgrid, 256, 0, stream>>>(x, Wq, bq, Wk, bk, Wv, bv, qb, kb, vpk);

    attn_kernel<<<(B_ * (T_ / 16)) / 4, 128, 0, stream>>>(
        (const unsigned int*)qb, (const unsigned int*)kb, (const unsigned int*)vpk,
        x, gamma, out);
}
